// ResidualAttentionBlock_4861902979522
// MI455X (gfx1250) — matrix operand order repaired and checked by reading, not compiled, not run on hardware
//
#include <hip/hip_runtime.h>
#include <hip/hip_bf16.h>

// ---------- types ----------
typedef __bf16 v16bf __attribute__((ext_vector_type(16)));
typedef float  v8f   __attribute__((ext_vector_type(8)));
typedef unsigned int u32x4 __attribute__((ext_vector_type(4)));

union FragU { u32x4 q[2]; v16bf v; };

// ---------- dims ----------
#define BS     2
#define NCTX   512
#define DMODEL 256
#define NHEAD  8
#define HDIM   32          // DMODEL / NHEAD
#define MTOT   (BS * NCTX) // 1024 rows

// workspace layout (bytes, all 256-aligned)
#define OFF_XBF     ((size_t)0)                       // x bf16            1024x256
#define OFF_WQKV    ((size_t)524288)                  // Wqkv^T bf16       768x256
#define OFF_WFC     ((size_t)917504)                  // Wfc^T bf16        1024x256
#define OFF_WPROJ   ((size_t)1441792)                 // Wproj^T bf16      256x1024
#define OFF_QKV     ((size_t)1966080)                 // qkv fp32          1024x768
#define OFF_XRES    ((size_t)5111808)                 // x+attn fp32       1024x256
#define OFF_XRESBF  ((size_t)6160384)                 // x+attn bf16       1024x256
#define OFF_HBF     ((size_t)6684672)                 // gelu(fc) bf16     1024x1024

__device__ __forceinline__ unsigned short f32_to_bf16(float f) {
    unsigned int u = __float_as_uint(f);
    unsigned int r = u + 0x7FFFu + ((u >> 16) & 1u);   // round-to-nearest-even
    return (unsigned short)(r >> 16);
}

// ---------- conversion kernels ----------
__global__ void cvt_bf16_kernel(const float* __restrict__ in,
                                unsigned short* __restrict__ out, int n) {
    int i = blockIdx.x * blockDim.x + threadIdx.x;
    if (i < n) out[i] = f32_to_bf16(in[i]);
}

// in: K x N row-major fp32  ->  out: N x K row-major bf16 (i.e. B^T, K contiguous)
__global__ void cvt_transpose_bf16_kernel(const float* __restrict__ in,
                                          unsigned short* __restrict__ out,
                                          int K, int N) {
    int i = blockIdx.x * blockDim.x + threadIdx.x;
    if (i >= K * N) return;
    int n = i / K, k = i - n * K;
    out[(size_t)n * K + k] = f32_to_bf16(in[(size_t)k * N + n]);
}

// ---------- WMMA fragment loaders (ISA 16-bit A 16x32 / B 32x16 layouts) ----------
__device__ __forceinline__ v16bf load_fragA(const unsigned short* __restrict__ A,
                                            int lda, int row0, int k0, int lane) {
    int hf  = lane >> 4;
    int row = row0 + (lane & 15);
    int ko  = k0 + (hf ? 8 : 0);                 // lanes 16-31 start at K+8
    const unsigned short* p = A + (size_t)row * lda + ko;
    FragU u;
    u.q[0] = *reinterpret_cast<const u32x4*>(p);       // K ko .. ko+7
    u.q[1] = *reinterpret_cast<const u32x4*>(p + 16);  // K ko+16 .. ko+23
    return u.v;
}

__device__ __forceinline__ v16bf load_fragB(const unsigned short* __restrict__ Bt,
                                            int ldb, int col0, int k0, int lane) {
    int hf  = lane >> 4;
    int col = col0 + (lane & 15);
    int ko  = k0 + (hf ? 8 : 0);                 // lanes 16-31 start at K+8
    const unsigned short* p = Bt + (size_t)col * ldb + ko;
    FragU u;
    u.q[0] = *reinterpret_cast<const u32x4*>(p);       // K ko .. ko+7
    u.q[1] = *reinterpret_cast<const u32x4*>(p + 16);  // K ko+16 .. ko+23
    return u.v;
}

// ---------- bf16 WMMA GEMM: C = A(MxK) * B^T(NxK) + bias, epilogues ----------
// EPI 0: outF = C + bias                     (qkv)
// EPI 1: outB = bf16(quick_gelu(C + bias))   (fc)
// EPI 2: outF = C + bias + res               (proj + residual)
template <int EPI>
__global__ void gemm_wmma_kernel(const unsigned short* __restrict__ A,
                                 const unsigned short* __restrict__ Bt,
                                 const float* __restrict__ bias,
                                 const float* __restrict__ res,
                                 float* __restrict__ outF,
                                 unsigned short* __restrict__ outB,
                                 int M, int N, int K) {
    int lane = threadIdx.x;                          // wave32
    int tile = blockIdx.x * 8 + threadIdx.y;         // 8 waves / block
    int tiles_m = M >> 5;
    if (tile >= tiles_m * (N >> 5)) return;          // wave-uniform guard
    int mt = tile % tiles_m;
    int nt = tile / tiles_m;
    int m0 = mt << 5, n0 = nt << 5;

    v8f acc[2][2] = {};
    for (int k0 = 0; k0 < K; k0 += 32) {
        v16bf a0 = load_fragA(A, K, m0,      k0, lane);
        v16bf a1 = load_fragA(A, K, m0 + 16, k0, lane);
        v16bf b0 = load_fragB(Bt, K, n0,      k0, lane);
        v16bf b1 = load_fragB(Bt, K, n0 + 16, k0, lane);
        acc[0][0] = __builtin_amdgcn_wmma_f32_16x16x32_bf16(false, a0, false, b0, (short)0, acc[0][0], false, false);
        acc[0][1] = __builtin_amdgcn_wmma_f32_16x16x32_bf16(false, a0, false, b1, (short)0, acc[0][1], false, false);
        acc[1][0] = __builtin_amdgcn_wmma_f32_16x16x32_bf16(false, a1, false, b0, (short)0, acc[1][0], false, false);
        acc[1][1] = __builtin_amdgcn_wmma_f32_16x16x32_bf16(false, a1, false, b1, (short)0, acc[1][1], false, false);
    }

    // C/D layout: lanes 0-15 -> M = r, lanes 16-31 -> M = 8 + r; N = lane & 15
    int hf = lane >> 4, l = lane & 15;
#pragma unroll
    for (int i = 0; i < 2; ++i) {
#pragma unroll
        for (int j = 0; j < 2; ++j) {
            int col = n0 + j * 16 + l;
            float bv = bias[col];
#pragma unroll
            for (int r = 0; r < 8; ++r) {
                int row = m0 + i * 16 + hf * 8 + r;
                size_t idx = (size_t)row * N + col;
                float v = acc[i][j][r] + bv;
                if (EPI == 0) {
                    outF[idx] = v;
                } else if (EPI == 1) {
                    float g = v * (1.0f / (1.0f + __expf(-1.702f * v)));
                    outB[idx] = f32_to_bf16(g);
                } else {
                    outF[idx] = v + res[idx];
                }
            }
        }
    }
}

// ---------- L1-distance attention + residual (fused softmax / weight*V) ----------
// grid: 128 blocks = b(2) * h(8) * tblock(8);  block: 32x8 (8 waves)
// LDS: kS[512*33] + vS[512*33] + qS[64*33] + wS[8*512]  = 160000 bytes
__global__ void attn_l1_kernel(const float* __restrict__ x,
                               const float* __restrict__ qkv,
                               float* __restrict__ xres,
                               unsigned short* __restrict__ xres_bf) {
    extern __shared__ float lds[];
    float* kS = lds;                  // 512*33
    float* vS = kS + 512 * 33;        // 512*33
    float* qS = vS + 512 * 33;        // 64*33
    float* wB = qS + 64 * 33;         // 8*512 (per-wave)

    int bid = blockIdx.x;
    int tb = bid & 7;
    int h  = (bid >> 3) & 7;
    int b  = bid >> 6;
    int tid  = threadIdx.y * 32 + threadIdx.x;
    int lane = threadIdx.x;
    int wave = threadIdx.y;
    int t0 = tb * 64;

    const float* qkvb = qkv + (size_t)b * NCTX * 768 + h * 96;
    for (int i = tid; i < NCTX * HDIM; i += 256) {
        int s = i >> 5, c = i & 31;
        const float* row = qkvb + (size_t)s * 768;
        kS[s * 33 + c] = row[32 + c];
        vS[s * 33 + c] = row[64 + c];
    }
    for (int i = tid; i < 64 * HDIM; i += 256) {
        int r = i >> 5, c = i & 31;
        qS[r * 33 + c] = qkvb[(size_t)(t0 + r) * 768 + c];
    }
    __syncthreads();

    const float scale = 0.1767766952966369f; // 1/sqrt(32)
    float* wS = wB + wave * 512;

    for (int tt = 0; tt < 8; ++tt) {
        int tl = wave * 8 + tt;     // row in block tile
        int t  = t0 + tl;           // row in sequence
        const float* qrow = qS + tl * 33;

        // weights for 16 s-values per lane (s = lane + 32*i), conflict-free (stride 33)
        float wloc[16];
        float wmax = 0.0f;
#pragma unroll
        for (int i = 0; i < 16; ++i) {
            int s = lane + 32 * i;
            const float* kp = kS + s * 33;
            float d = 0.0f;
#pragma unroll
            for (int c = 0; c < 32; ++c) d += fabsf(qrow[c] - kp[c]);
            float w = (s == t) ? 0.0f : 1.0f / (0.001f + d * scale);
            wloc[i] = w;
            wmax = fmaxf(wmax, w);
        }
        for (int m = 16; m >= 1; m >>= 1) wmax = fmaxf(wmax, __shfl_xor(wmax, m, 32));
        float lsum = 0.0f;
#pragma unroll
        for (int i = 0; i < 16; ++i) {
            float e = __expf(wloc[i] - wmax);
            wloc[i] = e;
            lsum += e;
        }
        for (int m = 16; m >= 1; m >>= 1) lsum += __shfl_xor(lsum, m, 32);
        float inv = 1.0f / lsum;

#pragma unroll
        for (int i = 0; i < 16; ++i) wS[lane + 32 * i] = wloc[i];

        // out[t, c=lane] = inv * sum_s w[s] * v[s, lane]; w broadcast, v stride-33
        float acc = 0.0f;
#pragma unroll 4
        for (int s = 0; s < 512; ++s) acc += wS[s] * vS[s * 33 + lane];
        acc *= inv;

        size_t oi = ((size_t)b * NCTX + t) * DMODEL + h * HDIM + lane;
        float r = x[oi] + acc;
        xres[oi] = r;
        xres_bf[oi] = f32_to_bf16(r);
    }
}

// ---------- launcher ----------
extern "C" void kernel_launch(void* const* d_in, const int* in_sizes, int n_in,
                              void* d_out, int out_size, void* d_ws, size_t ws_size,
                              hipStream_t stream) {
    const float* x     = (const float*)d_in[0];
    const float* Wqkv  = (const float*)d_in[1];
    const float* bqkv  = (const float*)d_in[2];
    const float* Wfc   = (const float*)d_in[3];
    const float* bfc   = (const float*)d_in[4];
    const float* Wproj = (const float*)d_in[5];
    const float* bproj = (const float*)d_in[6];

    char* ws = (char*)d_ws;
    unsigned short* xbf    = (unsigned short*)(ws + OFF_XBF);
    unsigned short* wqkvt  = (unsigned short*)(ws + OFF_WQKV);
    unsigned short* wfct   = (unsigned short*)(ws + OFF_WFC);
    unsigned short* wprojt = (unsigned short*)(ws + OFF_WPROJ);
    float*          qkvf   = (float*)(ws + OFF_QKV);
    float*          xresf  = (float*)(ws + OFF_XRES);
    unsigned short* xresbf = (unsigned short*)(ws + OFF_XRESBF);
    unsigned short* hbf    = (unsigned short*)(ws + OFF_HBF);

    dim3 gblk(32, 8);

    // fp32 -> bf16 conversions (weights transposed to NxK so WMMA B-fragments
    // become two contiguous 16B loads per lane)
    cvt_bf16_kernel<<<(MTOT * DMODEL + 255) / 256, 256, 0, stream>>>(x, xbf, MTOT * DMODEL);
    cvt_transpose_bf16_kernel<<<(DMODEL * 768 + 255) / 256, 256, 0, stream>>>(Wqkv, wqkvt, DMODEL, 768);
    cvt_transpose_bf16_kernel<<<(DMODEL * 1024 + 255) / 256, 256, 0, stream>>>(Wfc, wfct, DMODEL, 1024);
    cvt_transpose_bf16_kernel<<<(1024 * DMODEL + 255) / 256, 256, 0, stream>>>(Wproj, wprojt, 1024, DMODEL);

    // qkv = x @ Wqkv + b : M=1024 N=768 K=256 -> 32*24 = 768 wave-tiles -> 96 blocks
    gemm_wmma_kernel<0><<<96, gblk, 0, stream>>>(xbf, wqkvt, bqkv, nullptr, qkvf, nullptr,
                                                 MTOT, 768, DMODEL);

    // fused L1 attention + residual -> xres (fp32 + bf16)
    (void)hipFuncSetAttribute((const void*)attn_l1_kernel,
                              hipFuncAttributeMaxDynamicSharedMemorySize, 160000);
    attn_l1_kernel<<<128, gblk, 160000, stream>>>(x, qkvf, xresf, xresbf);

    // h = quick_gelu(xres @ Wfc + b) : M=1024 N=1024 K=256 -> 1024 tiles -> 128 blocks
    gemm_wmma_kernel<1><<<128, gblk, 0, stream>>>(xresbf, wfct, bfc, nullptr, nullptr, hbf,
                                                  MTOT, 1024, DMODEL);

    // out = xres + h @ Wproj + b : M=1024 N=256 K=1024 -> 256 tiles -> 32 blocks
    gemm_wmma_kernel<2><<<32, gblk, 0, stream>>>(hbf, wprojt, bproj, xresf, (float*)d_out, nullptr,
                                                 MTOT, DMODEL, 1024);
}
